// LSPE_MPGNN_51170240364732
// MI455X (gfx1250) — matrix operand order repaired and checked by reading, not compiled, not run on hardware
//
#include <hip/hip_runtime.h>

// ---------------------------------------------------------------------------
// LSPE-MPGNN for MI455X (gfx1250).
// All GEMMs run on v_wmma_f32_16x16x32_bf16 (f32 accumulate).
//  - B (weights) pre-packed in WMMA fragment-major order in global memory:
//    each lane loads its full v16bf operand with 2x global_load_b128 (L2-hot).
//  - A (features) gathered straight into fragments: per lane, two aligned
//    8-float K-runs per 32-K chunk, packed to bf16 pairs with
//    v_cvt_pk_bf16_f32. No LDS staging, no block barriers.
//  - segment_sum via hardware global_atomic_add_f32.
// ---------------------------------------------------------------------------

#define TPB 256              // 8 independent waves per block
#define ROWS_PER_WAVE 16
#define ROWS_PER_BLOCK 128
#define SLOT_DWORDS 2048     // fragment dwords per 64-wide K slot: 2*4*32*8

typedef __attribute__((ext_vector_type(16))) __bf16 v16bf;
typedef __attribute__((ext_vector_type(8)))  float  v8f;

union ABu { v16bf v; unsigned u[8]; };
union CFu { v8f  v; float    f[8]; };

__device__ __forceinline__ unsigned pack_bf16(float a, float b) {
  unsigned short ua = __builtin_bit_cast(unsigned short, (__bf16)a);
  unsigned short ub = __builtin_bit_cast(unsigned short, (__bf16)b);
  return (unsigned)ua | ((unsigned)ub << 16);
}

// ---------------------------------------------------------------------------
// Weight pre-pack into fragment-major order.
// Fragment dword index: {slot, kc, nt, lane, j} ; value packs B-matrix
// elements (K = slot*64 + kc*32 + (j>>2)*16 + half*8 + 2(j&3) {,+1}, N = nt*16 + (lane&15)),
// zero-padded past K (handles the 16-wide embedding weights).
// ---------------------------------------------------------------------------
__global__ void pack_frag_kernel(const float* __restrict__ W,
                                 unsigned* __restrict__ Fp,
                                 int K, int total) {
  const int i = blockIdx.x * blockDim.x + threadIdx.x;
  if (i >= total) return;
  const int j    = i & 7;
  const int lane = (i >> 3) & 31;
  const int nt   = (i >> 8) & 3;
  const int kc   = (i >> 10) & 1;
  const int slot = i >> 11;
  const int half = lane >> 4;
  const int n    = nt * 16 + (lane & 15);
  const int k0   = slot * 64 + kc * 32 + (j >> 2) * 16 + half * 8 + 2 * (j & 3);
  const float w0 = (k0     < K) ? W[(long)k0 * 64 + n]       : 0.f;
  const float w1 = (k0 + 1 < K) ? W[(long)(k0 + 1) * 64 + n] : 0.f;
  Fp[i] = pack_bf16(w0, w1);
}

// A fragment from a row pointer: two aligned 8-float runs at kb and kb+16.
// ncols is a multiple of 8, so runs are entirely valid or entirely zero.
__device__ __forceinline__ ABu build_A_runs(const float* __restrict__ rp,
                                            int kb, int ncols, bool rowvalid) {
  ABu a;
#pragma unroll
  for (int j = 0; j < 8; ++j) a.u[j] = 0u;
  if (rowvalid) {
    if (kb + 8 <= ncols) {
      const float4 f0 = *(const float4*)(rp + kb);
      const float4 f1 = *(const float4*)(rp + kb + 4);
      a.u[0] = pack_bf16(f0.x, f0.y); a.u[1] = pack_bf16(f0.z, f0.w);
      a.u[2] = pack_bf16(f1.x, f1.y); a.u[3] = pack_bf16(f1.z, f1.w);
    }
    if (kb + 24 <= ncols) {
      const float4 f2 = *(const float4*)(rp + kb + 16);
      const float4 f3 = *(const float4*)(rp + kb + 20);
      a.u[4] = pack_bf16(f2.x, f2.y); a.u[5] = pack_bf16(f2.z, f2.w);
      a.u[6] = pack_bf16(f3.x, f3.y); a.u[7] = pack_bf16(f3.z, f3.w);
    }
  }
  return a;
}

__device__ __forceinline__ void mma_4tiles(const ABu& a,
                                           const unsigned* __restrict__ Fp,
                                           int slot, int kc, int lane, v8f c[4]) {
#pragma unroll
  for (int nt = 0; nt < 4; ++nt) {
    const unsigned* p = Fp + (((slot * 2 + kc) * 4 + nt) * 32 + lane) * 8;
    const v16bf b = *(const v16bf*)p;   // 2x global_load_b128, L2-hot
    c[nt] = __builtin_amdgcn_wmma_f32_16x16x32_bf16(
        false, a.v, false, b, (short)0, c[nt], false, false);
  }
}

// ---------------------------------------------------------------------------
// Node-side GEMM: out[M,64] = concat(src0[:,colofs0:+ncols0], src1[...]) @ W + b
// ---------------------------------------------------------------------------
__global__ __launch_bounds__(TPB) void node_gemm_kernel(
    const float* __restrict__ src0, long stride0, int colofs0, int ncols0,
    const float* __restrict__ src1, long stride1, int colofs1, int ncols1,
    const unsigned* __restrict__ Fp, const float* __restrict__ bias,
    float* __restrict__ out, int M) {
  const int tid  = threadIdx.x;
  const int lane = tid & 31;
  const int wv   = tid >> 5;
  const int row_base = blockIdx.x * ROWS_PER_BLOCK + wv * ROWS_PER_WAVE;

  const int m = lane & 15, half = lane >> 4, nl = lane & 15;
  const int row_m = row_base + m;
  const bool rv = row_m < M;

  v8f c[4] = {};
  const int nslots = (ncols1 > 0) ? 2 : 1;
  for (int s = 0; s < nslots; ++s) {
    const float* src = s ? src1 : src0;
    const long  strd = s ? stride1 : stride0;
    const int   cofs = s ? colofs1 : colofs0;
    const int   ncol = s ? ncols1 : ncols0;
    const float* rp = src + (long)row_m * strd + cofs;
    const int kcmax = (ncol > 32) ? 2 : 1;   // skip all-zero K chunks
    for (int kc = 0; kc < kcmax; ++kc) {
      const ABu a = build_A_runs(rp, kc * 32 + half * 8, ncol, rv);
      mma_4tiles(a, Fp, s, kc, lane, c);
    }
  }

  if (row_base + ROWS_PER_WAVE <= M) {        // full tile: unguarded stores
#pragma unroll
    for (int nt = 0; nt < 4; ++nt) {
      CFu cf; cf.v = c[nt];
      const int n = nt * 16 + nl;
      const float bn = bias[n];
#pragma unroll
      for (int r = 0; r < 8; ++r)
        out[(long)(row_base + half * 8 + r) * 64 + n] = cf.f[r] + bn;
    }
  } else {
#pragma unroll
    for (int nt = 0; nt < 4; ++nt) {
      CFu cf; cf.v = c[nt];
      const int n = nt * 16 + nl;
      const float bn = bias[n];
#pragma unroll
      for (int r = 0; r < 8; ++r) {
        const int row = row_base + half * 8 + r;
        if (row < M) out[(long)row * 64 + n] = cf.f[r] + bn;
      }
    }
  }
}

// ---------------------------------------------------------------------------
// Edge-side GEMM with gather on the A side.
//   slot mode: 0 = direct edge row, 1 = gather via send[], 2 = gather via rec[]
//   scatter:   0 -> out[edge,64] = result + bias
//              1 -> atomic add of (result + bias) into out[rec[edge],64]
// ---------------------------------------------------------------------------
__global__ __launch_bounds__(TPB) void edge_gemm_kernel(
    const float* __restrict__ s0, const float* __restrict__ s1,
    const float* __restrict__ s2, const float* __restrict__ s3,
    const float* __restrict__ s4,
    int m0, int m1, int m2, int m3, int m4, int nslots,
    const int* __restrict__ send, const int* __restrict__ rec,
    const unsigned* __restrict__ Fp, const float* __restrict__ bias,
    float* __restrict__ out, int scatter, int E) {
  const int tid  = threadIdx.x;
  const int lane = tid & 31;
  const int wv   = tid >> 5;
  const int row_base = blockIdx.x * ROWS_PER_BLOCK + wv * ROWS_PER_WAVE;

  // Speculative prefetch of upcoming edge indices (global_prefetch_b8).
  {
    const int pf = row_base + ROWS_PER_BLOCK * 8 + lane;
    if (pf < E) {
      __builtin_prefetch(send + pf, 0, 1);
      __builtin_prefetch(rec + pf, 0, 1);
    }
  }

  const float* srcs[5]  = {s0, s1, s2, s3, s4};
  const int    modes[5] = {m0, m1, m2, m3, m4};

  const int m = lane & 15, half = lane >> 4, nl = lane & 15;
  const int row_m = row_base + m;
  const bool rv = row_m < E;
  const int sIdx = rv ? send[row_m] : 0;     // gather indices loaded once
  const int rIdx = rv ? rec[row_m]  : 0;

  v8f c[4] = {};
  for (int s = 0; s < nslots; ++s) {
    const int mode = modes[s];
    const long base = (mode == 1) ? (long)sIdx : (mode == 2) ? (long)rIdx : (long)row_m;
    const float* rp = srcs[s] + base * 64;
#pragma unroll
    for (int kc = 0; kc < 2; ++kc) {
      const ABu a = build_A_runs(rp, kc * 32 + half * 8, 64, rv);
      mma_4tiles(a, Fp, s, kc, lane, c);
    }
  }

  const bool full = (row_base + ROWS_PER_WAVE) <= E;
  if (scatter) {
#pragma unroll
    for (int nt = 0; nt < 4; ++nt) {
      CFu cf; cf.v = c[nt];
      const int n = nt * 16 + nl;
      const float bn = bias[n];
#pragma unroll
      for (int r = 0; r < 8; ++r) {
        const int srow = half * 8 + r;
        const int dst_node = __shfl(rIdx, srow, 32);   // lane srow holds rec[row_base+srow]
        if (full || (row_base + srow) < E) {
          float* dst = out + (long)dst_node * 64 + n;
          (void)__hip_atomic_fetch_add(dst, cf.f[r] + bn, __ATOMIC_RELAXED,
                                       __HIP_MEMORY_SCOPE_AGENT);
        }
      }
    }
  } else if (full) {
#pragma unroll
    for (int nt = 0; nt < 4; ++nt) {
      CFu cf; cf.v = c[nt];
      const int n = nt * 16 + nl;
      const float bn = bias[n];
#pragma unroll
      for (int r = 0; r < 8; ++r)
        out[(long)(row_base + half * 8 + r) * 64 + n] = cf.f[r] + bn;
    }
  } else {
#pragma unroll
    for (int nt = 0; nt < 4; ++nt) {
      CFu cf; cf.v = c[nt];
      const int n = nt * 16 + nl;
      const float bn = bias[n];
#pragma unroll
      for (int r = 0; r < 8; ++r) {
        const int row = row_base + half * 8 + r;
        if (row < E) out[(long)row * 64 + n] = cf.f[r] + bn;
      }
    }
  }
}

// ---------------------------------------------------------------------------
// Host orchestration
// ---------------------------------------------------------------------------
extern "C" void kernel_launch(void* const* d_in, const int* in_sizes, int n_in,
                              void* d_out, int out_size, void* d_ws, size_t ws_size,
                              hipStream_t stream) {
  const float* h_in = (const float*)d_in[0];   // [N,128]
  const float* e_in = (const float*)d_in[1];   // [E,16]
  const float* p_in = (const float*)d_in[2];   // [N,16]
  const int*   ei   = (const int*)d_in[3];     // [2,E]
  const float* he_W = (const float*)d_in[4];   const float* he_b = (const float*)d_in[5];
  const float* ee_W = (const float*)d_in[6];   const float* ee_b = (const float*)d_in[7];
  const float* pe_W = (const float*)d_in[8];   const float* pe_b = (const float*)d_in[9];
  const float* hm_W = (const float*)d_in[10];  const float* hm_b = (const float*)d_in[11];
  const float* hu_W = (const float*)d_in[12];  const float* hu_b = (const float*)d_in[13];
  const float* eu_W = (const float*)d_in[14];  const float* eu_b = (const float*)d_in[15];
  const float* pm_W = (const float*)d_in[16];  const float* pm_b = (const float*)d_in[17];
  const float* pu_W = (const float*)d_in[18];  const float* pu_b = (const float*)d_in[19];

  const int N = in_sizes[0] / 128;
  const int E = in_sizes[1] / 16;
  const int L = 4;
  const int* send = ei;
  const int* rec  = ei + E;

  // ---- workspace carve-up ----
  unsigned* wsu = (unsigned*)d_ws;
  size_t uo = 0;
  auto alloc_u = [&](size_t nslots) { unsigned* r = wsu + uo; uo += nslots * SLOT_DWORDS; return r; };
  unsigned* he_f = alloc_u(2);   // K=128
  unsigned* ee_f = alloc_u(1);   // K=16 (padded)
  unsigned* pe_f = alloc_u(1);
  unsigned *hm_f[4], *hu_f[4], *eu_f[4], *pm_f[4], *pu_f[4];
  for (int i = 0; i < L; ++i) {
    hm_f[i] = alloc_u(5);        // K=320
    hu_f[i] = alloc_u(2);        // K=128
    eu_f[i] = alloc_u(3);        // K=192
    pm_f[i] = alloc_u(3);        // K=192
    pu_f[i] = alloc_u(2);        // K=128
  }
  float* wsf = (float*)(wsu + uo);
  size_t fo = 0;
  auto alloc_f = [&](size_t n) { float* r = wsf + fo; fo += n; return r; };
  float* hbuf0 = alloc_f((size_t)N * 64);
  float* hbuf1 = alloc_f((size_t)N * 64);
  float* pbuf0 = alloc_f((size_t)N * 64);
  float* pbuf1 = alloc_f((size_t)N * 64);
  float* agg   = alloc_f((size_t)N * 64);
  float* ebuf  = alloc_f((size_t)E * 64);

  // ---- pack all weights into fragment-major bf16 layout ----
  auto pack = [&](const float* W, unsigned* Fp, int K, int nslots) {
    const int total = nslots * SLOT_DWORDS;
    pack_frag_kernel<<<(total + 255) / 256, 256, 0, stream>>>(W, Fp, K, total);
  };
  pack(he_W, he_f, 128, 2);
  pack(ee_W, ee_f, 16, 1);
  pack(pe_W, pe_f, 16, 1);
  for (int i = 0; i < L; ++i) {
    pack(hm_W + (size_t)i * 320 * 64, hm_f[i], 320, 5);
    pack(hu_W + (size_t)i * 128 * 64, hu_f[i], 128, 2);
    pack(eu_W + (size_t)i * 192 * 64, eu_f[i], 192, 3);
    pack(pm_W + (size_t)i * 192 * 64, pm_f[i], 192, 3);
    pack(pu_W + (size_t)i * 128 * 64, pu_f[i], 128, 2);
  }

  const int grid_n = (N + ROWS_PER_BLOCK - 1) / ROWS_PER_BLOCK;
  const int grid_e = (E + ROWS_PER_BLOCK - 1) / ROWS_PER_BLOCK;

  // ---- input embeddings ----
  node_gemm_kernel<<<grid_n, TPB, 0, stream>>>(h_in, 128, 0, 64, h_in, 128, 64, 64,
                                               he_f, he_b, hbuf0, N);
  node_gemm_kernel<<<grid_n, TPB, 0, stream>>>(p_in, 16, 0, 16, nullptr, 0, 0, 0,
                                               pe_f, pe_b, pbuf0, N);
  node_gemm_kernel<<<grid_e, TPB, 0, stream>>>(e_in, 16, 0, 16, nullptr, 0, 0, 0,
                                               ee_f, ee_b, ebuf, E);

  // ---- L message-passing layers ----
  float* hc = hbuf0;
  float* pc = pbuf0;
  for (int i = 0; i < L; ++i) {
    float* hn = (i == L - 1) ? (float*)d_out                  : ((i & 1) ? hbuf0 : hbuf1);
    float* pn = (i == L - 1) ? (float*)d_out + (size_t)N * 64 : ((i & 1) ? pbuf0 : pbuf1);

    // h_agg = segment_sum( [h_s, p_s, h_r, p_r, e] @ hm_W + hm_b , rec )
    hipMemsetAsync(agg, 0, (size_t)N * 64 * sizeof(float), stream);
    edge_gemm_kernel<<<grid_e, TPB, 0, stream>>>(
        hc, pc, hc, pc, ebuf, 1, 1, 2, 2, 0, 5, send, rec,
        hm_f[i], hm_b + (size_t)i * 64, agg, 1, E);

    // h = [h, h_agg] @ hu_W + hu_b
    node_gemm_kernel<<<grid_n, TPB, 0, stream>>>(hc, 64, 0, 64, agg, 64, 0, 64,
                                                 hu_f[i], hu_b + (size_t)i * 64, hn, N);

    // e = [h_s, h_r, e] @ eu_W + eu_b   (in place; row-local)
    edge_gemm_kernel<<<grid_e, TPB, 0, stream>>>(
        hn, hn, ebuf, nullptr, nullptr, 1, 2, 0, 0, 0, 3, send, rec,
        eu_f[i], eu_b + (size_t)i * 64, ebuf, 0, E);

    // p_agg = segment_sum( [p_s, p_r, e] @ pm_W + pm_b , rec )
    hipMemsetAsync(agg, 0, (size_t)N * 64 * sizeof(float), stream);
    edge_gemm_kernel<<<grid_e, TPB, 0, stream>>>(
        pc, pc, ebuf, nullptr, nullptr, 1, 2, 0, 0, 0, 3, send, rec,
        pm_f[i], pm_b + (size_t)i * 64, agg, 1, E);

    // p = [p, p_agg] @ pu_W + pu_b
    node_gemm_kernel<<<grid_n, TPB, 0, stream>>>(pc, 64, 0, 64, agg, 64, 0, 64,
                                                 pu_f[i], pu_b + (size_t)i * 64, pn, N);

    hc = hn;
    pc = pn;
  }
}
